// MetaLayer_2267742732914
// MI455X (gfx1250) — matrix operand order repaired and checked by reading, not compiled, not run on hardware
//
#include <hip/hip_runtime.h>
#include <stdint.h>

// ---------------------------------------------------------------------------
// MetaLayer (hetero GNN) for MI455X / gfx1250.
// All GEMMs routed through v_wmma_f32_16x16x32_bf16 (fp32 accumulate).
// ---------------------------------------------------------------------------

typedef __bf16 bf16_t;
typedef __attribute__((ext_vector_type(16))) __bf16 v16bf;
typedef __attribute__((ext_vector_type(8)))  float  v8f;

#define DN 64
#define DE 32
#define DH 64

__device__ __forceinline__ float relu_f(float x) { return x > 0.f ? x : 0.f; }

__device__ __forceinline__ void atomic_add_f32(float* p, float v) {
  // lowers to global_atomic_add_f32 on gfx1250
  __hip_atomic_fetch_add(p, v, __ATOMIC_RELAXED, __HIP_MEMORY_SCOPE_AGENT);
}

__device__ __forceinline__ v8f wmma_bf16(v16bf a, v16bf b, v8f c) {
  return __builtin_amdgcn_wmma_f32_16x16x32_bf16(
      /*neg_a=*/false, a, /*neg_b=*/false, b,
      /*c_mod=*/(short)0, c, /*reuse_a=*/false, /*reuse_b=*/false);
}

// ---------------------------------------------------------------------------
// Zero-fill workspace aggregates.
// ---------------------------------------------------------------------------
__global__ void zero_kernel(float* __restrict__ p, size_t n) {
  size_t i = (size_t)blockIdx.x * blockDim.x + threadIdx.x;
  if (i < n) p[i] = 0.f;
}

// ---------------------------------------------------------------------------
// Pack a row-major fp32 weight [K, Nn] into WMMA B-fragment order (bf16).
// Fragment (kc, nt) covers K rows [kc*32, kc*32+32), cols [nt*16, nt*16+16).
// B layout (16-bit, 32x16): lanes 0-15 -> K = kc*32 + 0..7, 16..23 (elems 0..15),
//                           lanes 16-31 -> K = kc*32 + 8..15, 24..31, N = lane&15.
// One block (32 lanes) per fragment; each lane writes 16 contiguous bf16.
// ---------------------------------------------------------------------------
__global__ void pack_w_kernel(const float* __restrict__ W, bf16_t* __restrict__ dst,
                              int K, int Nn) {
  int frag = blockIdx.x;
  int lane = threadIdx.x;
  int nTiles = Nn >> 4;
  int kc = frag / nTiles, nt = frag % nTiles;
  int half = lane >> 4, l = lane & 15;
  int n  = nt * 16 + l;
  int k0 = kc * 32 + half * 8;
  bf16_t* o = dst + (size_t)frag * 512 + (size_t)lane * 16;
#pragma unroll
  for (int j = 0; j < 16; ++j)
    o[j] = (bf16_t)W[(size_t)(k0 + j + (j >= 8 ? 8 : 0)) * Nn + n];
}

// ---------------------------------------------------------------------------
// Edge MLP: e = relu(relu([x_s | x_d | ea] @ W1 + b1) @ W2 + b2)
// plus scatter-add of e into aggR[row] and aggC[col].
// 8 waves / block, 16 edges / wave.
// ---------------------------------------------------------------------------
__global__ __launch_bounds__(256) void edge_mlp_kernel(
    const float* __restrict__ xs, const float* __restrict__ xd,
    const float* __restrict__ ea, const int* __restrict__ ei,
    const bf16_t* __restrict__ W1p, const float* __restrict__ b1,
    const bf16_t* __restrict__ W2p, const float* __restrict__ b2,
    float* __restrict__ e_out, float* __restrict__ aggR, float* __restrict__ aggC,
    int E) {
  // h rows per wave, padded to 72 bf16 (144 B) to spread LDS banks.
  __shared__ bf16_t h_lds[8][16][72];
  __shared__ int    idx_lds[8][16][2];

  int tid  = threadIdx.x;
  int wave = tid >> 5, lane = tid & 31;
  int half = lane >> 4, l = lane & 15;
  int tile = (blockIdx.x * 8 + wave) * 16;

  int m  = tile + l;
  int mc = (m < E) ? m : (E - 1);          // clamped for gathers
  int r  = ei[mc];
  int c  = ei[(size_t)E + mc];
  const float* xsrow = xs + (size_t)r * DN;
  const float* xdrow = xd + (size_t)c * DN;
  const float* earow = ea + (size_t)mc * DE;

  if (lane < 16) { idx_lds[wave][l][0] = r; idx_lds[wave][l][1] = c; }

  // ---- layer 1: [16,160] @ [160,64] ----
  v8f acc0 = 0, acc1 = 0, acc2 = 0, acc3 = 0;
#pragma unroll
  for (int kc = 0; kc < 5; ++kc) {
    // A fragment: elems 0..7 = f[g1..g1+7], elems 8..15 = f[g1+16..g1+23]
    int g1 = kc * 32 + half * 8;
    int g2 = g1 + 16;
    const float* p1 = (g1 < 64) ? (xsrow + g1)
                    : (g1 < 128) ? (xdrow + (g1 - 64)) : (earow + (g1 - 128));
    const float* p2 = (g2 < 64) ? (xsrow + g2)
                    : (g2 < 128) ? (xdrow + (g2 - 64)) : (earow + (g2 - 128));
    float4 q0 = ((const float4*)p1)[0], q1 = ((const float4*)p1)[1];
    float4 q2 = ((const float4*)p2)[0], q3 = ((const float4*)p2)[1];
    v16bf a;
    a[0] = (bf16_t)q0.x; a[1] = (bf16_t)q0.y; a[2]  = (bf16_t)q0.z; a[3]  = (bf16_t)q0.w;
    a[4] = (bf16_t)q1.x; a[5] = (bf16_t)q1.y; a[6]  = (bf16_t)q1.z; a[7]  = (bf16_t)q1.w;
    a[8] = (bf16_t)q2.x; a[9] = (bf16_t)q2.y; a[10] = (bf16_t)q2.z; a[11] = (bf16_t)q2.w;
    a[12]= (bf16_t)q3.x; a[13]= (bf16_t)q3.y; a[14] = (bf16_t)q3.z; a[15] = (bf16_t)q3.w;

    const bf16_t* wb = W1p + (size_t)(kc * 4) * 512 + (size_t)lane * 16;
    acc0 = wmma_bf16(a, *(const v16bf*)(wb + 0 * 512 * 32 / 32), acc0);
    acc1 = wmma_bf16(a, *(const v16bf*)(wb + 1 * 512), acc1);
    acc2 = wmma_bf16(a, *(const v16bf*)(wb + 2 * 512), acc2);
    acc3 = wmma_bf16(a, *(const v16bf*)(wb + 3 * 512), acc3);
  }

  // bias + relu -> LDS (bf16), D layout: VGPR i holds row M = i + 8*half, col nt*16+l
  {
    v8f* accs[4] = { &acc0, &acc1, &acc2, &acc3 };
#pragma unroll
    for (int nt = 0; nt < 4; ++nt) {
      float bias = b1[nt * 16 + l];
#pragma unroll
      for (int i = 0; i < 8; ++i) {
        float v = relu_f((*accs[nt])[i] + bias);
        h_lds[wave][i + half * 8][nt * 16 + l] = (bf16_t)v;
      }
    }
  }
  __syncthreads();

  // ---- layer 2: [16,64] @ [64,32] ----
  v8f d0 = 0, d1 = 0;
  const bf16_t* hrow = &h_lds[wave][l][0];
#pragma unroll
  for (int kc = 0; kc < 2; ++kc) {
    int g1 = kc * 32 + half * 8;
    int g2 = g1 + 16;
    v16bf a;
#pragma unroll
    for (int j = 0; j < 8; ++j) { a[j] = hrow[g1 + j]; a[8 + j] = hrow[g2 + j]; }
    const bf16_t* wb = W2p + (size_t)(kc * 2) * 512 + (size_t)lane * 16;
    d0 = wmma_bf16(a, *(const v16bf*)(wb + 0 * 512), d0);
    d1 = wmma_bf16(a, *(const v16bf*)(wb + 1 * 512), d1);
  }

  // bias + relu, store e, scatter-add into aggregates
  {
    v8f* ds[2] = { &d0, &d1 };
#pragma unroll
    for (int nt = 0; nt < 2; ++nt) {
      float bias = b2[nt * 16 + l];
#pragma unroll
      for (int i = 0; i < 8; ++i) {
        float v  = relu_f((*ds[nt])[i] + bias);
        int   mi = tile + i + half * 8;       // edge row of this element
        if (mi < E) {
          int er = idx_lds[wave][i + half * 8][0];
          int ec = idx_lds[wave][i + half * 8][1];
          e_out[(size_t)mi * DE + nt * 16 + l] = v;
          atomic_add_f32(&aggR[(size_t)er * DE + nt * 16 + l], v);
          atomic_add_f32(&aggC[(size_t)ec * DE + nt * 16 + l], v);
        }
      }
    }
  }
}

// ---------------------------------------------------------------------------
// Node MLP: new_x = relu([x | agg] @ Wn + bn), [N,96] @ [96,64]
// ---------------------------------------------------------------------------
__global__ __launch_bounds__(256) void node_mlp_kernel(
    const float* __restrict__ x, const float* __restrict__ agg,
    const bf16_t* __restrict__ Wnp, const float* __restrict__ bn,
    float* __restrict__ out, int Nn) {
  int tid  = threadIdx.x;
  int wave = tid >> 5, lane = tid & 31;
  int half = lane >> 4, l = lane & 15;
  int tile = (blockIdx.x * 8 + wave) * 16;
  int m  = tile + l;
  int mc = (m < Nn) ? m : (Nn - 1);
  const float* xrow = x   + (size_t)mc * DN;
  const float* arow = agg + (size_t)mc * DE;

  v8f acc0 = 0, acc1 = 0, acc2 = 0, acc3 = 0;
#pragma unroll
  for (int kc = 0; kc < 3; ++kc) {
    int g1 = kc * 32 + half * 8;
    int g2 = g1 + 16;
    const float* p1 = (g1 < 64) ? (xrow + g1) : (arow + (g1 - 64));
    const float* p2 = (g2 < 64) ? (xrow + g2) : (arow + (g2 - 64));
    float4 q0 = ((const float4*)p1)[0], q1 = ((const float4*)p1)[1];
    float4 q2 = ((const float4*)p2)[0], q3 = ((const float4*)p2)[1];
    v16bf a;
    a[0] = (bf16_t)q0.x; a[1] = (bf16_t)q0.y; a[2]  = (bf16_t)q0.z; a[3]  = (bf16_t)q0.w;
    a[4] = (bf16_t)q1.x; a[5] = (bf16_t)q1.y; a[6]  = (bf16_t)q1.z; a[7]  = (bf16_t)q1.w;
    a[8] = (bf16_t)q2.x; a[9] = (bf16_t)q2.y; a[10] = (bf16_t)q2.z; a[11] = (bf16_t)q2.w;
    a[12]= (bf16_t)q3.x; a[13]= (bf16_t)q3.y; a[14] = (bf16_t)q3.z; a[15] = (bf16_t)q3.w;

    const bf16_t* wb = Wnp + (size_t)(kc * 4) * 512 + (size_t)lane * 16;
    acc0 = wmma_bf16(a, *(const v16bf*)(wb + 0 * 512), acc0);
    acc1 = wmma_bf16(a, *(const v16bf*)(wb + 1 * 512), acc1);
    acc2 = wmma_bf16(a, *(const v16bf*)(wb + 2 * 512), acc2);
    acc3 = wmma_bf16(a, *(const v16bf*)(wb + 3 * 512), acc3);
  }

  v8f* accs[4] = { &acc0, &acc1, &acc2, &acc3 };
#pragma unroll
  for (int nt = 0; nt < 4; ++nt) {
    float bias = bn[nt * 16 + l];
#pragma unroll
    for (int i = 0; i < 8; ++i) {
      float v  = relu_f((*accs[nt])[i] + bias);
      int   mi = tile + i + half * 8;
      if (mi < Nn) out[(size_t)mi * DN + nt * 16 + l] = v;
    }
  }
}

// ---------------------------------------------------------------------------
// Host launcher.
// Input order (setup_inputs dict order):
//  0 x_a, 1 x_b,
//  2 ea_aa, 3 W1_aa, 4 b1_aa, 5 W2_aa, 6 b2_aa, 7 ei_aa,
//  8 ea_ab, 9 W1_ab,10 b1_ab,11 W2_ab,12 b2_ab,13 ei_ab,
// 14 ea_bb,15 W1_bb,16 b1_bb,17 W2_bb,18 b2_bb,19 ei_bb,
// 20 Wn_a,21 bn_a,22 Wn_b,23 bn_b
// Output: [new_x_a (N*64) | new_x_b (N*64) | e_aa | e_ab | e_bb (E*32 each)]
// ---------------------------------------------------------------------------
extern "C" void kernel_launch(void* const* d_in, const int* in_sizes, int n_in,
                              void* d_out, int out_size, void* d_ws, size_t ws_size,
                              hipStream_t stream) {
  const float* x_a = (const float*)d_in[0];
  const float* x_b = (const float*)d_in[1];
  const float* ea[3] = {(const float*)d_in[2],  (const float*)d_in[8],  (const float*)d_in[14]};
  const float* W1[3] = {(const float*)d_in[3],  (const float*)d_in[9],  (const float*)d_in[15]};
  const float* b1[3] = {(const float*)d_in[4],  (const float*)d_in[10], (const float*)d_in[16]};
  const float* W2[3] = {(const float*)d_in[5],  (const float*)d_in[11], (const float*)d_in[17]};
  const float* b2[3] = {(const float*)d_in[6],  (const float*)d_in[12], (const float*)d_in[18]};
  const int*   ei[3] = {(const int*)d_in[7],    (const int*)d_in[13],   (const int*)d_in[19]};
  const float* Wn[2] = {(const float*)d_in[20], (const float*)d_in[22]};
  const float* bn[2] = {(const float*)d_in[21], (const float*)d_in[23]};

  int Nn = in_sizes[0] / DN;     // nodes per type
  int E  = in_sizes[2] / DE;     // edges per edge type

  // ---- workspace layout ----
  char*  ws   = (char*)d_ws;
  float* aggA = (float*)ws;
  float* aggB = aggA + (size_t)Nn * DE;
  uintptr_t wraw = (uintptr_t)(aggB + (size_t)Nn * DE);
  bf16_t* cur = (bf16_t*)((wraw + 31) & ~(uintptr_t)31);
  bf16_t *W1p[3], *W2p[3], *Wnp[2];
  for (int k = 0; k < 3; ++k) { W1p[k] = cur; cur += 20 * 512; }   // 5 Kchunks x 4 Ntiles
  for (int k = 0; k < 3; ++k) { W2p[k] = cur; cur += 4  * 512; }   // 2 x 2
  for (int t = 0; t < 2; ++t) { Wnp[t] = cur; cur += 12 * 512; }   // 3 x 4

  // ---- zero aggregates (aggA and aggB are contiguous) ----
  size_t zn = 2 * (size_t)Nn * DE;
  zero_kernel<<<(unsigned)((zn + 255) / 256), 256, 0, stream>>>(aggA, zn);

  // ---- pack weights to bf16 WMMA fragment order ----
  for (int k = 0; k < 3; ++k) {
    pack_w_kernel<<<20, 32, 0, stream>>>(W1[k], W1p[k], 2 * DN + DE, DH);
    pack_w_kernel<<<4,  32, 0, stream>>>(W2[k], W2p[k], DH, DE);
  }
  for (int t = 0; t < 2; ++t)
    pack_w_kernel<<<12, 32, 0, stream>>>(Wn[t], Wnp[t], DN + DE, DN);

  // ---- output layout ----
  float* out = (float*)d_out;
  float* nxa = out;
  float* nxb = out + (size_t)Nn * DN;
  float* e_out[3];
  e_out[0] = nxb + (size_t)Nn * DN;
  e_out[1] = e_out[0] + (size_t)E * DE;
  e_out[2] = e_out[1] + (size_t)E * DE;

  const float* xs[3]  = {x_a, x_a, x_b};
  const float* xd[3]  = {x_a, x_b, x_b};
  float*       agR[3] = {aggA, aggA, aggB};
  float*       agC[3] = {aggA, aggB, aggB};

  int eblocks = (E + 127) / 128;
  for (int k = 0; k < 3; ++k) {
    edge_mlp_kernel<<<eblocks, 256, 0, stream>>>(
        xs[k], xd[k], ea[k], ei[k], W1p[k], b1[k], W2p[k], b2[k],
        e_out[k], agR[k], agC[k], E);
  }

  int nblocks = (Nn + 127) / 128;
  node_mlp_kernel<<<nblocks, 256, 0, stream>>>(x_a, aggA, Wnp[0], bn[0], nxa, Nn);
  node_mlp_kernel<<<nblocks, 256, 0, stream>>>(x_b, aggB, Wnp[1], bn[1], nxb, Nn);
}
